// scBERT_54571854463417
// MI455X (gfx1250) — matrix operand order repaired and checked by reading, not compiled, not run on hardware
//
#include <hip/hip_runtime.h>
#include <hip/hip_bf16.h>

#define DEPTH_  4
#define BATCH_  4
#define SEQ_    8192
#define DIM_    256
#define HEADS_  8
#define DHEAD_  32
#define INNER_  256
#define NB_     110
#define NBP_    128
#define FF_     1024
#define NTOK_   (BATCH_*SEQ_)          // 32768
#define NROWS_  (BATCH_*HEADS_*SEQ_)   // 262144
#define CTX_SPLIT_ 8

typedef __attribute__((ext_vector_type(16))) _Float16 v16h;
typedef __attribute__((ext_vector_type(8)))  _Float16 v8h;
typedef __attribute__((ext_vector_type(8)))  float    v8f;

// ---------------------------------------------------------------- utilities
static __device__ __forceinline__ v8f wmma32(v16h a, v16h b, v8f c) {
  return __builtin_amdgcn_wmma_f32_16x16x32_f16(false, a, false, b, (short)0, c,
                                                false, false);
}

static __device__ __forceinline__ float waveSum(float v) {
  #pragma unroll
  for (int m = 16; m >= 1; m >>= 1) v += __shfl_xor(v, m, 32);
  return v;
}
static __device__ __forceinline__ float waveMax(float v) {
  #pragma unroll
  for (int m = 16; m >= 1; m >>= 1) v = fmaxf(v, __shfl_xor(v, m, 32));
  return v;
}

// order-preserving float <-> uint encoding for atomicMax
static __device__ __forceinline__ unsigned encF(float f) {
  int i = __float_as_int(f);
  return (i >= 0) ? ((unsigned)i | 0x80000000u) : (unsigned)(~i);
}
static __device__ __forceinline__ float decF(unsigned u) {
  int i = (u & 0x80000000u) ? (int)(u & 0x7fffffffu) : ~(int)u;
  return __int_as_float(i);
}

// A fragment 16x32 f16 from row-major [M x K] (ld even, 16B-aligned rows):
// lane -> row (lane&15); lanes<16 hold K {0..15}, lanes>=16 K {16..31}.
// Per lane these are two contiguous 16-byte chunks -> two b128 loads.
static __device__ __forceinline__ v16h load_frag_a(const _Float16* __restrict__ src, int ld) {
  const int lane = threadIdx.x & 31;
  const _Float16* p = src + (size_t)(lane & 15) * ld + ((lane & 16) ? 16 : 0);
  v8h lo = *(const v8h*)p;          // K +0..7
  v8h hi = *(const v8h*)(p + 8);    // K +8..15
  v16h f;
  #pragma unroll
  for (int j = 0; j < 8; ++j) { f[j] = lo[j]; f[8 + j] = hi[j]; }
  return f;
}

// B fragment 32x16 f16 from pre-swizzled tile [32 lanes][16 f16] -> 2 b128 loads
static __device__ __forceinline__ v16h load_frag_b_swz(const _Float16* __restrict__ tile) {
  const int lane = threadIdx.x & 31;
  return *(const v16h*)(tile + lane * 16);
}

// ---------------------------------------------------------------- GEMM
// B operand is swizzled tiles: [K/32][ntilesN][32][16] f16 (512 elems / tile)
// MODE 1: gelu(tanh approx) -> f16 [M,ldout]
// MODE 2: f32 out[row*ldout+col] += val   (residual accumulate)
// MODE 3: f16 store to QKV head layout [B,H,N,DHEAD]
// MODE 4: f16 store to swizzled B-fragment layout for V: [bh][n/32][d/16][32][16]
// block = 256 thr (8 waves), block tile 512x64, wave tile 64x64 (4 M-substrips):
// per K-step 4 B frags are reused by 4 A frags -> 16 WMMA per 16 b128 loads.
template <int MODE>
__global__ void gemm_kernel(const _Float16* __restrict__ A, int lda,
                            const _Float16* __restrict__ Bswz, int ntilesN,
                            const float* __restrict__ bias,
                            void* __restrict__ outp, int ldout, int K) {
  const int lane = threadIdx.x & 31;
  const int wave = threadIdx.x >> 5;
  const int mBase = blockIdx.y * 512 + wave * 64;
  const int nt0 = blockIdx.x * 4;
  v8f acc[4][4] = {};
  for (int k = 0; k < K; k += 32) {
    if (k + 32 < K) __builtin_prefetch(A + (size_t)mBase * lda + k + 32);
    v16h b[4];
    const _Float16* bb = Bswz + ((size_t)(k >> 5) * ntilesN + nt0) * 512;
    #pragma unroll
    for (int nt = 0; nt < 4; ++nt) b[nt] = load_frag_b_swz(bb + nt * 512);
    #pragma unroll
    for (int ms = 0; ms < 4; ++ms) {
      v16h a = load_frag_a(A + (size_t)(mBase + ms * 16) * lda + k, lda);
      #pragma unroll
      for (int nt = 0; nt < 4; ++nt)
        acc[ms][nt] = wmma32(a, b[nt], acc[ms][nt]);
    }
  }
  const int rOff = (lane & 16) ? 8 : 0;
  #pragma unroll
  for (int ms = 0; ms < 4; ++ms) {
    #pragma unroll
    for (int nt = 0; nt < 4; ++nt) {
      int col = (nt0 + nt) * 16 + (lane & 15);
      float bv = bias[col];
      #pragma unroll
      for (int v = 0; v < 8; ++v) {
        int row = mBase + ms * 16 + v + rOff;
        float val = acc[ms][nt][v] + bv;
        if (MODE == 1) {
          float u = val;
          float t = 0.7978845608f * (u + 0.044715f * u * u * u);
          val = 0.5f * u * (1.0f + tanhf(t));
          ((_Float16*)outp)[(size_t)row * ldout + col] = (_Float16)val;
        } else if (MODE == 2) {
          float* o = (float*)outp;
          o[(size_t)row * ldout + col] += val;
        } else if (MODE == 3) {  // q/k head layout [bh][n][32]
          int head = col >> 5, d = col & 31;
          int b_ = row >> 13, n_ = row & (SEQ_ - 1);
          ((_Float16*)outp)[((size_t)((b_ * HEADS_ + head) << 13) + n_) * DHEAD_ + d] =
              (_Float16)val;
        } else {  // MODE 4: V directly in swizzled ctx-B layout
          int head = col >> 5, d = col & 31;
          int b_ = row >> 13, n_ = row & (SEQ_ - 1);
          int bh = b_ * HEADS_ + head;
          int ktv = n_ >> 5, kin = n_ & 31;
          int ntv = d >> 4, cin = d & 15;
          int lanev = cin + (kin & 16);
          int jv = kin & 15;
          ((_Float16*)outp)[(((size_t)bh * (SEQ_ / 32) + ktv) * 2 + ntv) * 512 +
                            lanev * 16 + jv] = (_Float16)val;
        }
      }
    }
  }
}

// ---------------------------------------------------------------- FAVOR+ features
// pass 1 for keys: global max over (n, nb) per (b,h). projT pre-swizzled: 8 tiles.
__global__ void feat_max_kernel(const _Float16* __restrict__ xh,
                                const _Float16* __restrict__ projT,
                                unsigned* __restrict__ kmax) {
  const int lane = threadIdx.x & 31;
  const int wave = threadIdx.x >> 5;
  const int r0 = blockIdx.x * 64 + wave * 16;
  const int bh = r0 >> 13;
  v16h a = load_frag_a(xh + (size_t)r0 * DHEAD_, DHEAD_);
  float lm = -3.4e38f;
  #pragma unroll
  for (int nt = 0; nt < 7; ++nt) {  // cols 0..111 cover the 110 valid features
    v8f z = {};
    v8f acc = wmma32(a, load_frag_b_swz(projT + nt * 512), z);
    int col = nt * 16 + (lane & 15);
    if (col < NB_) {
      #pragma unroll
      for (int v = 0; v < 8; ++v) lm = fmaxf(lm, acc[v]);
    }
  }
  float wm = waveMax(lm);
  if (lane == 0) atomicMax(&kmax[bh], encF(wm));
}

// xd via WMMA -> LDS -> exp epilogue (diag via wave reduction of x^2)
// isQuery=1: per-row max, store row-major [row][128]
// isQuery=0: global max from kmax, store TRANSPOSED [bh][128][SEQ] (feeds ctx A-frags)
__global__ void feat_kernel(const _Float16* __restrict__ xh,
                            const _Float16* __restrict__ projT,
                            const unsigned* __restrict__ kmax,
                            _Float16* __restrict__ outp, int isQuery) {
  __shared__ float tile[4][16][NBP_];
  const int lane = threadIdx.x & 31;
  const int wave = threadIdx.x >> 5;
  const int r0 = blockIdx.x * 64 + wave * 16;
  const int bh = r0 >> 13;
  v16h a = load_frag_a(xh + (size_t)r0 * DHEAD_, DHEAD_);
  const int rOff = (lane & 16) ? 8 : 0;
  #pragma unroll
  for (int nt = 0; nt < 8; ++nt) {
    v8f z = {};
    v8f acc = wmma32(a, load_frag_b_swz(projT + nt * 512), z);
    #pragma unroll
    for (int v = 0; v < 8; ++v)
      tile[wave][v + rOff][nt * 16 + (lane & 15)] = acc[v];
  }
  __syncthreads();
  const float halfdn2 = 0.0883883476f;   // 0.5 * DHEAD^-0.5
  const float ratio = 0.0953462589f;     // NB^-0.5
  float gmx = isQuery ? 0.f : decF(kmax[bh]);
  for (int r = 0; r < 16; ++r) {
    int row = r0 + r;
    float xv = (float)xh[(size_t)row * DHEAD_ + lane];
    float diag = halfdn2 * waveSum(xv * xv);
    float mx = gmx;
    if (isQuery) {
      float lm = -3.4e38f;
      #pragma unroll
      for (int j = 0; j < 4; ++j) {
        int c = lane + 32 * j;
        if (c < NB_) lm = fmaxf(lm, tile[wave][r][c]);
      }
      mx = waveMax(lm);
    }
    #pragma unroll
    for (int j = 0; j < 4; ++j) {
      int c = lane + 32 * j;
      float val = (c < NB_)
          ? ratio * (expf(tile[wave][r][c] - diag - mx) + 1e-4f)
          : 0.f;
      if (isQuery)
        outp[(size_t)row * NBP_ + c] = (_Float16)val;
      else
        outp[((size_t)(bh * NBP_ + c)) * SEQ_ + (row & (SEQ_ - 1))] = (_Float16)val;
    }
  }
}

// ---------------------------------------------------------------- ctx = kp^T @ [v | 1]
// kpT: [bh][128][SEQ] f16 (A-frags are plain row-major loads, ld=SEQ)
// vswz: [bh][SEQ/32][2][32][16] swizzled B tiles
// split-K over sequence: grid (BH, CTX_SPLIT), partials atomically added into
// f32 ctxf [bh][128][48]
__global__ void ctx_kernel(const _Float16* __restrict__ kpT,
                           const _Float16* __restrict__ vswz,
                           float* __restrict__ ctxf) {
  const int lane = threadIdx.x & 31;
  const int wave = threadIdx.x >> 5;  // 0..7
  const int bh = blockIdx.x;
  const int m0 = wave * 16;
  const int kBeg = blockIdx.y * (SEQ_ / CTX_SPLIT_);
  const int kEnd = kBeg + (SEQ_ / CTX_SPLIT_);
  v16h bones;
  {
    _Float16 o = ((lane & 15) == 0) ? (_Float16)1.0f : (_Float16)0.0f;
    #pragma unroll
    for (int i = 0; i < 16; ++i) bones[i] = o;
  }
  const _Float16* aBase = kpT + ((size_t)(bh * NBP_ + m0)) * SEQ_;
  const _Float16* vBase = vswz + (size_t)bh * (SEQ_ / 32) * 2 * 512;
  v8f a0 = {}, a1 = {}, a2 = {};
  for (int k0 = kBeg; k0 < kEnd; k0 += 32) {
    __builtin_prefetch(aBase + k0 + 32);
    v16h a = load_frag_a(aBase + k0, SEQ_);
    const _Float16* vb = vBase + (size_t)(k0 >> 5) * 2 * 512;
    a0 = wmma32(a, load_frag_b_swz(vb), a0);
    a1 = wmma32(a, load_frag_b_swz(vb + 512), a1);
    a2 = wmma32(a, bones, a2);
  }
  const int rOff = (lane & 16) ? 8 : 0;
  const int c = lane & 15;
  #pragma unroll
  for (int v = 0; v < 8; ++v) {
    int m = m0 + v + rOff;
    float* dst = ctxf + ((size_t)bh * NBP_ + m) * 48 + c;
    atomicAdd(dst, a0[v]);
    atomicAdd(dst + 16, a1[v]);
    atomicAdd(dst + 32, a2[v]);
  }
}

// pack f32 ctx [bh][128][48] -> swizzled f16 B tiles [bh][4][3][32][16]
__global__ void ctx_pack_kernel(const float* __restrict__ ctxf,
                                _Float16* __restrict__ ctxp) {
  int i = blockIdx.x * blockDim.x + threadIdx.x;  // 32*128*48
  if (i >= BATCH_ * HEADS_ * NBP_ * 48) return;
  int bh = i / (NBP_ * 48);
  int r = i % (NBP_ * 48);
  int m = r / 48;
  int c = r % 48;
  int ctkt = m >> 5, kin = m & 31;
  int nt = c >> 4, cin = c & 15;
  int lanep = cin + (kin & 16);
  int j = kin & 15;
  ctxp[(((size_t)bh * 4 + ctkt) * 3 + nt) * 512 + lanep * 16 + j] = (_Float16)ctxf[i];
}

// ---------------------------------------------------------------- o = (qp @ ctxp) / col32
__global__ void combine_kernel(const _Float16* __restrict__ qp,
                               const _Float16* __restrict__ ctxp,
                               _Float16* __restrict__ attn) {
  __shared__ float tile[4][16][48];
  const int lane = threadIdx.x & 31;
  const int wave = threadIdx.x >> 5;
  const int r0 = blockIdx.x * 64 + wave * 16;
  const int bh = r0 >> 13;
  v8f acc[3] = {};
  const _Float16* cb = ctxp + (size_t)bh * 4 * 3 * 512;
  #pragma unroll
  for (int kk = 0; kk < 4; ++kk) {
    v16h a = load_frag_a(qp + (size_t)r0 * NBP_ + kk * 32, NBP_);
    #pragma unroll
    for (int nt = 0; nt < 3; ++nt)
      acc[nt] = wmma32(a, load_frag_b_swz(cb + ((size_t)kk * 3 + nt) * 512), acc[nt]);
  }
  const int rOff = (lane & 16) ? 8 : 0;
  #pragma unroll
  for (int nt = 0; nt < 3; ++nt)
    #pragma unroll
    for (int v = 0; v < 8; ++v)
      tile[wave][v + rOff][nt * 16 + (lane & 15)] = acc[nt][v];
  __syncthreads();
  int b_ = bh >> 3, h_ = bh & 7;
  for (int r = 0; r < 16; ++r) {
    int row = r0 + r;
    int n_ = row & (SEQ_ - 1);
    float inv = 1.0f / tile[wave][r][32];
    float val = tile[wave][r][lane] * inv;
    attn[((size_t)(b_ * SEQ_ + n_)) * INNER_ + h_ * DHEAD_ + lane] = (_Float16)val;
  }
}

// ---------------------------------------------------------------- layernorm (f32 -> f16)
__global__ void ln_kernel(const float* __restrict__ h, const float* __restrict__ g,
                          const float* __restrict__ bb, _Float16* __restrict__ y) {
  __shared__ float red[8];
  const int row = blockIdx.x, t = threadIdx.x;  // 256 threads
  float x = h[(size_t)row * DIM_ + t];
  float s = waveSum(x);
  if ((t & 31) == 0) red[t >> 5] = s;
  __syncthreads();
  float tot = 0.f;
  #pragma unroll
  for (int i = 0; i < 8; ++i) tot += red[i];
  float mean = tot * (1.0f / DIM_);
  __syncthreads();
  float d = x - mean;
  float s2 = waveSum(d * d);
  if ((t & 31) == 0) red[t >> 5] = s2;
  __syncthreads();
  float tv = 0.f;
  #pragma unroll
  for (int i = 0; i < 8; ++i) tv += red[i];
  float inv = rsqrtf(tv * (1.0f / DIM_) + 1e-5f);
  y[(size_t)row * DIM_ + t] = (_Float16)(d * inv * g[t] + bb[t]);
}

// ---------------------------------------------------------------- small helpers
__global__ void copy_f32(const float* __restrict__ s, float* __restrict__ d, int n) {
  int i = blockIdx.x * blockDim.x + threadIdx.x;
  if (i < n) d[i] = s[i];
}

// f32 row-major weights [DEPTH][K][N] -> swizzled f16 B tiles [DEPTH][K/32][N/16][32][16]
__global__ void swz_w_kernel(const float* __restrict__ W, _Float16* __restrict__ dst,
                             int K, int N, int total) {
  int i = blockIdx.x * blockDim.x + threadIdx.x;
  if (i >= total) return;
  int layer = i / (K * N);
  int r = i % (K * N);
  int j = r & 15;
  int lane = (r >> 4) & 31;
  int t = r >> 9;
  int ntiles = N >> 4;
  int nt = t % ntiles;
  int kt = t / ntiles;
  int k = kt * 32 + ((lane & 16) ? 16 : 0) + j;
  int col = nt * 16 + (lane & 15);
  dst[i] = (_Float16)W[(size_t)layer * K * N + (size_t)k * N + col];
}

// proj [DEPTH][NB][32] f32 -> swizzled projT tiles [DEPTH][8][32][16] f16, dn folded, pad 0
__global__ void projT_kernel(const float* __restrict__ proj, _Float16* __restrict__ dst) {
  int i = blockIdx.x * blockDim.x + threadIdx.x;  // DEPTH * 4096
  if (i >= DEPTH_ * 4096) return;
  int layer = i >> 12;
  int r = i & 4095;
  int j = r & 15;
  int lane = (r >> 4) & 31;
  int nt = r >> 9;  // 0..7
  int k = ((lane & 16) ? 16 : 0) + j;
  int m = nt * 16 + (lane & 15);
  float v = (m < NB_) ? proj[(size_t)layer * NB_ * DHEAD_ + m * DHEAD_ + k] * 0.4204482076f
                      : 0.f;  // dn = 32^-0.25
  dst[i] = (_Float16)v;
}

// ---------------------------------------------------------------- host driver
extern "C" void kernel_launch(void* const* d_in, const int* in_sizes, int n_in,
                              void* d_out, int out_size, void* d_ws, size_t ws_size,
                              hipStream_t stream) {
  (void)in_sizes; (void)n_in; (void)out_size; (void)ws_size;
  const float* x    = (const float*)d_in[0];
  const float* proj = (const float*)d_in[1];
  const float* ln1g = (const float*)d_in[2];
  const float* ln1b = (const float*)d_in[3];
  const float* Wq   = (const float*)d_in[4];
  const float* bq   = (const float*)d_in[5];
  const float* Wk   = (const float*)d_in[6];
  const float* bk   = (const float*)d_in[7];
  const float* Wv   = (const float*)d_in[8];
  const float* bv   = (const float*)d_in[9];
  const float* Wo   = (const float*)d_in[10];
  const float* bo   = (const float*)d_in[11];
  const float* ln2g = (const float*)d_in[12];
  const float* ln2b = (const float*)d_in[13];
  const float* Wf1  = (const float*)d_in[14];
  const float* bf1  = (const float*)d_in[15];
  const float* Wf2  = (const float*)d_in[16];
  const float* bf2  = (const float*)d_in[17];
  float* h = (float*)d_out;

  char* ws = (char*)d_ws;
  size_t off = 0;
  auto alloc = [&](size_t bytes) -> void* {
    void* p = ws + off;
    off += (bytes + 255) & ~(size_t)255;
    return p;
  };
  _Float16* y16    = (_Float16*)alloc((size_t)NTOK_ * DIM_ * 2);
  _Float16* q16    = (_Float16*)alloc((size_t)NROWS_ * DHEAD_ * 2);
  _Float16* k16    = (_Float16*)alloc((size_t)NROWS_ * DHEAD_ * 2);
  _Float16* vswz   = (_Float16*)alloc((size_t)NROWS_ * DHEAD_ * 2);
  _Float16* attn16 = (_Float16*)alloc((size_t)NTOK_ * INNER_ * 2);
  float*    ctxf   = (float*)alloc((size_t)BATCH_ * HEADS_ * NBP_ * 48 * 4);
  _Float16* ctxp   = (_Float16*)alloc((size_t)BATCH_ * HEADS_ * 4 * 3 * 512 * 2);
  unsigned* kmax   = (unsigned*)alloc(256);
  _Float16* wq16   = (_Float16*)alloc((size_t)DEPTH_ * DIM_ * INNER_ * 2);
  _Float16* wk16   = (_Float16*)alloc((size_t)DEPTH_ * DIM_ * INNER_ * 2);
  _Float16* wv16   = (_Float16*)alloc((size_t)DEPTH_ * DIM_ * INNER_ * 2);
  _Float16* wo16   = (_Float16*)alloc((size_t)DEPTH_ * INNER_ * DIM_ * 2);
  _Float16* wf1_16 = (_Float16*)alloc((size_t)DEPTH_ * DIM_ * FF_ * 2);
  _Float16* wf2_16 = (_Float16*)alloc((size_t)DEPTH_ * FF_ * DIM_ * 2);
  _Float16* pT16   = (_Float16*)alloc((size_t)DEPTH_ * 4096 * 2);
  _Float16* qp16   = (_Float16*)alloc((size_t)NROWS_ * NBP_ * 2);
  _Float16* kpT16  = (_Float16*)alloc((size_t)NROWS_ * NBP_ * 2);
  _Float16* g16    = qp16;  // FFN hidden aliases qp region (same size: 32768*1024)

  const int T = 256;
  copy_f32<<<(NTOK_ * DIM_ + T - 1) / T, T, 0, stream>>>(x, h, NTOK_ * DIM_);
  {
    int tq = DEPTH_ * DIM_ * INNER_;
    swz_w_kernel<<<(tq + T - 1) / T, T, 0, stream>>>(Wq, wq16, DIM_, INNER_, tq);
    swz_w_kernel<<<(tq + T - 1) / T, T, 0, stream>>>(Wk, wk16, DIM_, INNER_, tq);
    swz_w_kernel<<<(tq + T - 1) / T, T, 0, stream>>>(Wv, wv16, DIM_, INNER_, tq);
    swz_w_kernel<<<(tq + T - 1) / T, T, 0, stream>>>(Wo, wo16, INNER_, DIM_, tq);
    int tf = DEPTH_ * DIM_ * FF_;
    swz_w_kernel<<<(tf + T - 1) / T, T, 0, stream>>>(Wf1, wf1_16, DIM_, FF_, tf);
    swz_w_kernel<<<(tf + T - 1) / T, T, 0, stream>>>(Wf2, wf2_16, FF_, DIM_, tf);
  }
  projT_kernel<<<(DEPTH_ * 4096 + T - 1) / T, T, 0, stream>>>(proj, pT16);

  for (int i = 0; i < DEPTH_; ++i) {
    ln_kernel<<<NTOK_, DIM_, 0, stream>>>(h, ln1g + i * DIM_, ln1b + i * DIM_, y16);

    dim3 gQ(INNER_ / 64, NTOK_ / 512);
    gemm_kernel<3><<<gQ, 256, 0, stream>>>(y16, DIM_, wq16 + (size_t)i * DIM_ * INNER_,
                                           INNER_ / 16, bq + i * INNER_, q16, 0, DIM_);
    gemm_kernel<3><<<gQ, 256, 0, stream>>>(y16, DIM_, wk16 + (size_t)i * DIM_ * INNER_,
                                           INNER_ / 16, bk + i * INNER_, k16, 0, DIM_);
    gemm_kernel<4><<<gQ, 256, 0, stream>>>(y16, DIM_, wv16 + (size_t)i * DIM_ * INNER_,
                                           INNER_ / 16, bv + i * INNER_, vswz, 0, DIM_);

    hipMemsetAsync(kmax, 0, 32 * sizeof(unsigned), stream);
    const _Float16* pT = pT16 + (size_t)i * 4096;
    feat_max_kernel<<<NROWS_ / 64, 128, 0, stream>>>(k16, pT, kmax);
    feat_kernel<<<NROWS_ / 64, 128, 0, stream>>>(q16, pT, kmax, qp16, 1);
    feat_kernel<<<NROWS_ / 64, 128, 0, stream>>>(k16, pT, kmax, kpT16, 0);

    hipMemsetAsync(ctxf, 0, (size_t)BATCH_ * HEADS_ * NBP_ * 48 * 4, stream);
    ctx_kernel<<<dim3(BATCH_ * HEADS_, CTX_SPLIT_), 256, 0, stream>>>(kpT16, vswz, ctxf);
    {
      int tc = BATCH_ * HEADS_ * NBP_ * 48;
      ctx_pack_kernel<<<(tc + T - 1) / T, T, 0, stream>>>(ctxf, ctxp);
    }
    combine_kernel<<<NROWS_ / 64, 128, 0, stream>>>(qp16, ctxp, attn16);

    gemm_kernel<2><<<dim3(DIM_ / 64, NTOK_ / 512), 256, 0, stream>>>(
        attn16, INNER_, wo16 + (size_t)i * INNER_ * DIM_, DIM_ / 16, bo + i * DIM_, h,
        DIM_, INNER_);

    ln_kernel<<<NTOK_, DIM_, 0, stream>>>(h, ln2g + i * DIM_, ln2b + i * DIM_, y16);
    gemm_kernel<1><<<dim3(FF_ / 64, NTOK_ / 512), 256, 0, stream>>>(
        y16, DIM_, wf1_16 + (size_t)i * DIM_ * FF_, FF_ / 16, bf1 + i * FF_, g16, FF_, DIM_);
    gemm_kernel<2><<<dim3(DIM_ / 64, NTOK_ / 512), 256, 0, stream>>>(
        g16, FF_, wf2_16 + (size_t)i * FF_ * DIM_, DIM_ / 16, bf2 + i * DIM_, h, DIM_, FF_);
  }
}
